// G_Res_Net_27797028339962
// MI455X (gfx1250) — compile-verified, matrix operand order checked by reading, not run on hardware
//
#include <hip/hip_runtime.h>
#include <hip/hip_bf16.h>

typedef __attribute__((ext_vector_type(16))) __bf16 v16bf;
typedef __attribute__((ext_vector_type(8)))  __bf16 v8bf;
typedef __attribute__((ext_vector_type(4)))  __bf16 v4bf;
typedef __attribute__((ext_vector_type(8)))  float  v8f;

#define NND 8192
#define INF 256
#define HIDC 192
#define SLEN 64
#define KCHUNK 128
#define LDSLD  136   // padded LDS row stride (bf16 elems) -> 272B rows, conflict-free

static __device__ __forceinline__ v16bf cat16(v8bf lo, v8bf hi) {
    return __builtin_shufflevector(lo, hi, 0,1,2,3,4,5,6,7,8,9,10,11,12,13,14,15);
}

// ---------------- converters ----------------
__global__ void conv_bf16_4(const float* __restrict__ src, __bf16* __restrict__ dst) {
    size_t idx = (size_t)blockIdx.x * blockDim.x + threadIdx.x;
    const float4 f = ((const float4*)src)[idx];
    v4bf o; o[0] = (__bf16)f.x; o[1] = (__bf16)f.y; o[2] = (__bf16)f.z; o[3] = (__bf16)f.w;
    *(v4bf*)(dst + idx * 4) = o;
}

// W (K x Nn, row-major fp32) -> W_T (Nn x K, row-major bf16)
__global__ void transpose_w(const float* __restrict__ W, __bf16* __restrict__ WT, int K, int Nn) {
    int idx = blockIdx.x * blockDim.x + threadIdx.x;
    if (idx >= K * Nn) return;
    int n = idx / K, k = idx % K;
    WT[(size_t)n * K + k] = (__bf16)W[(size_t)k * Nn + n];
}

// 12 stacked HID x HID weights
__global__ void transpose_wmid(const float* __restrict__ W, __bf16* __restrict__ WT) {
    int idx = blockIdx.x * blockDim.x + threadIdx.x;
    int l = idx / (HIDC * HIDC);
    int r = idx % (HIDC * HIDC);
    int n = r / HIDC, k = r % HIDC;
    WT[(size_t)l * HIDC * HIDC + (size_t)n * HIDC + k] =
        (__bf16)W[(size_t)l * HIDC * HIDC + (size_t)k * HIDC + n];
}

// ---------------- support = x @ W (WMMA), fused split epilogue ----------------
// cols [0,64):  write pre-bias/pre-relu, transposed into sT (stride NND)
// cols [64,192): relu(v + b) -> nxt[row*192 + col]
__global__ void feat_gemm(const __bf16* __restrict__ x, int ldx, int ksteps,
                          const __bf16* __restrict__ wT, int ldw,
                          const float* __restrict__ bias,
                          __bf16* __restrict__ sT, __bf16* __restrict__ nxt) {
    const int wave  = threadIdx.x >> 5;
    const int lane  = threadIdx.x & 31;
    const int l15   = lane & 15;
    const int hi    = (lane >> 4) & 1;
    const int ntile = blockIdx.y * 4 + wave;
    const int mBase = blockIdx.x * 16;
    const int row   = mBase + l15;
    const int ncol  = ntile * 16 + l15;

    const __bf16* ap = x  + (size_t)row  * ldx + hi * 8;
    const __bf16* bp = wT + (size_t)ncol * ldw + hi * 8;

    v8f c = {};
    #pragma unroll 2
    for (int kk = 0; kk < ksteps; ++kk) {
        const int k0 = kk * 32;
        v16bf a = cat16(*(const v8bf*)(ap + k0), *(const v8bf*)(ap + k0 + 16));
        v16bf b = cat16(*(const v8bf*)(bp + k0), *(const v8bf*)(bp + k0 + 16));
        c = __builtin_amdgcn_wmma_f32_16x16x32_bf16(false, a, false, b, (short)0, c, false, false);
    }

    if (ntile < 4) {
        #pragma unroll
        for (int r = 0; r < 8; ++r) {
            int rr = mBase + r + hi * 8;
            sT[(size_t)ncol * NND + rr] = (__bf16)c[r];
        }
    } else {
        const float bv = bias[ncol];
        #pragma unroll
        for (int r = 0; r < 8; ++r) {
            int rr = mBase + r + hi * 8;
            float v = c[r] + bv;
            v = v > 0.f ? v : 0.f;
            nxt[(size_t)rr * HIDC + ncol] = (__bf16)v;
        }
    }
}

// ---------------- out[:, :64] = relu(adjB @ S + b) (WMMA, LDS-staged B) ----------------
__global__ void adj_gemm(const __bf16* __restrict__ adjB,
                         const __bf16* __restrict__ sT,
                         const float* __restrict__ bias,
                         __bf16* __restrict__ nxt) {
    __shared__ __bf16 lb[SLEN * LDSLD];   // ~17 KB
    const int tid   = threadIdx.x;        // 0..127
    const int wave  = tid >> 5;           // 0..3 -> M tile within block
    const int lane  = tid & 31;
    const int l15   = lane & 15;
    const int hi    = (lane >> 4) & 1;
    const int mBase = (blockIdx.x * 4 + wave) * 16;
    const int row   = mBase + l15;

    const __bf16* ap = adjB + (size_t)row * NND + hi * 8;

    v8f acc[4] = {};

    for (int kb = 0; kb < NND; kb += KCHUNK) {
        __syncthreads();
        // stage 64 x 128 bf16 of S_T: 1024 v8bf units over 128 threads
        #pragma unroll
        for (int u = 0; u < 8; ++u) {
            int unit = tid * 8 + u;
            int n  = unit >> 4;           // 16 units per n-row
            int ko = (unit & 15) * 8;
            *(v8bf*)(&lb[n * LDSLD + ko]) = *(const v8bf*)(sT + (size_t)n * NND + kb + ko);
        }
        __syncthreads();
        #pragma unroll
        for (int kc = 0; kc < KCHUNK; kc += 32) {
            v16bf a = cat16(*(const v8bf*)(ap + kb + kc), *(const v8bf*)(ap + kb + kc + 16));
            #pragma unroll
            for (int nt = 0; nt < 4; ++nt) {
                const __bf16* lp = &lb[(nt * 16 + l15) * LDSLD + kc + hi * 8];
                v16bf b = cat16(*(const v8bf*)lp, *(const v8bf*)(lp + 16));
                acc[nt] = __builtin_amdgcn_wmma_f32_16x16x32_bf16(false, a, false, b,
                                                                  (short)0, acc[nt], false, false);
            }
        }
    }

    #pragma unroll
    for (int nt = 0; nt < 4; ++nt) {
        const int j  = nt * 16 + l15;
        const float bv = bias[j];
        #pragma unroll
        for (int r = 0; r < 8; ++r) {
            int rr = mBase + r + hi * 8;
            float v = acc[nt][r] + bv;
            v = v > 0.f ? v : 0.f;
            nxt[(size_t)rr * HIDC + j] = (__bf16)v;
        }
    }
}

// ---------------- residual merges ----------------
__global__ void res_init(const __bf16* __restrict__ featB, const __bf16* __restrict__ x,
                         __bf16* __restrict__ fbuf) {
    int idx = blockIdx.x * blockDim.x + threadIdx.x;   // over NND*HIDC
    int i = idx / HIDC, j = idx % HIDC;
    float v = ((float)featB[(size_t)i * INF + j] + (float)x[idx]) * 0.5f;
    fbuf[idx] = (__bf16)v;
}
__global__ void res_acc(const __bf16* __restrict__ x, __bf16* __restrict__ fbuf) {
    int idx = blockIdx.x * blockDim.x + threadIdx.x;
    fbuf[idx] = (__bf16)(((float)fbuf[idx] + (float)x[idx]) * 0.5f);
}

// ---------------- output layer (3 cols, side_len=2, no relu) ----------------
__global__ void out_gemm(const __bf16* __restrict__ fbuf, const float* __restrict__ Wout,
                         float* __restrict__ sup3) {
    int idx = blockIdx.x * blockDim.x + threadIdx.x;
    if (idx >= NND * 3) return;
    int i = idx / 3, j = idx % 3;
    const __bf16* fr = fbuf + (size_t)i * HIDC;
    float s = 0.f;
    for (int k = 0; k < HIDC; ++k) s += (float)fr[k] * Wout[k * 3 + j];
    sup3[idx] = s;
}
__global__ void out_agg(const __bf16* __restrict__ adjB, const float* __restrict__ sup3,
                        const float* __restrict__ bout, float* __restrict__ outp) {
    __shared__ float s0[256], s1[256];
    const int i = blockIdx.x, t = threadIdx.x;
    float a0 = 0.f, a1 = 0.f;
    const __bf16* ar = adjB + (size_t)i * NND;
    for (int k = t; k < NND; k += 256) {
        float a = (float)ar[k];
        a0 += a * sup3[k * 3 + 0];
        a1 += a * sup3[k * 3 + 1];
    }
    s0[t] = a0; s1[t] = a1;
    __syncthreads();
    for (int off = 128; off > 0; off >>= 1) {
        if (t < off) { s0[t] += s0[t + off]; s1[t] += s1[t + off]; }
        __syncthreads();
    }
    if (t == 0) {
        outp[i * 3 + 0] = s0[0] + bout[0];
        outp[i * 3 + 1] = s1[0] + bout[1];
        outp[i * 3 + 2] = sup3[i * 3 + 2] + bout[2];
    }
}
__global__ void feat_out(const __bf16* __restrict__ fbuf, float* __restrict__ outp) {
    int idx = blockIdx.x * blockDim.x + threadIdx.x;
    outp[idx] = (float)fbuf[idx];
}

extern "C" void kernel_launch(void* const* d_in, const int* in_sizes, int n_in,
                              void* d_out, int out_size, void* d_ws, size_t ws_size,
                              hipStream_t stream) {
    const float* features = (const float*)d_in[0];
    const float* adj      = (const float*)d_in[1];
    const float* W1       = (const float*)d_in[2];
    const float* b1       = (const float*)d_in[3];
    const float* Wmid     = (const float*)d_in[4];
    const float* bmid     = (const float*)d_in[5];
    const float* Wout     = (const float*)d_in[6];
    const float* bout     = (const float*)d_in[7];
    float* outp = (float*)d_out;

    char* p = (char*)d_ws;
    __bf16* adjB  = (__bf16*)p; p += (size_t)NND * NND * 2;     // 128 MB (L2-resident)
    __bf16* featB = (__bf16*)p; p += (size_t)NND * INF * 2;
    __bf16* W1T   = (__bf16*)p; p += (size_t)HIDC * INF * 2;
    __bf16* WmT   = (__bf16*)p; p += (size_t)12 * HIDC * HIDC * 2;
    __bf16* sT    = (__bf16*)p; p += (size_t)SLEN * NND * 2;
    __bf16* xA    = (__bf16*)p; p += (size_t)NND * HIDC * 2;
    __bf16* xB    = (__bf16*)p; p += (size_t)NND * HIDC * 2;
    __bf16* fbuf  = (__bf16*)p; p += (size_t)NND * HIDC * 2;
    float*  sup3  = (float*)p;  p += (size_t)NND * 3 * 4;

    // one-shot conversions (recomputed each call: deterministic)
    conv_bf16_4<<<(size_t)NND * NND / 4 / 256, 256, 0, stream>>>(adj, adjB);
    conv_bf16_4<<<(size_t)NND * INF / 4 / 256, 256, 0, stream>>>(features, featB);
    transpose_w<<<(INF * HIDC + 255) / 256, 256, 0, stream>>>(W1, W1T, INF, HIDC);
    transpose_wmid<<<12 * HIDC * HIDC / 256, 256, 0, stream>>>(Wmid, WmT);

    const dim3 fg(NND / 16, HIDC / 64);     // (512, 3) x 128 threads
    const int  ag = NND / 16 / 4;           // 128 blocks x 128 threads
    const int  ew = NND * HIDC / 256;       // 6144 elementwise blocks

    // gc1: features -> xA
    feat_gemm<<<fg, 128, 0, stream>>>(featB, INF, INF / 32, W1T, INF, b1, sT, xA);
    adj_gemm<<<ag, 128, 0, stream>>>(adjB, sT, b1, xA);
    // gc2: xA -> xB
    feat_gemm<<<fg, 128, 0, stream>>>(xA, HIDC, HIDC / 32, WmT, HIDC, bmid, sT, xB);
    adj_gemm<<<ag, 128, 0, stream>>>(adjB, sT, bmid, xB);
    res_init<<<ew, 256, 0, stream>>>(featB, xB, fbuf);

    // 5 residual pairs: W_mid[1..10]
    for (int i = 1; i <= 9; i += 2) {
        const __bf16* wa = WmT + (size_t)i * HIDC * HIDC;
        const __bf16* wb = WmT + (size_t)(i + 1) * HIDC * HIDC;
        const float*  ba = bmid + i * HIDC;
        const float*  bb = bmid + (i + 1) * HIDC;
        feat_gemm<<<fg, 128, 0, stream>>>(fbuf, HIDC, HIDC / 32, wa, HIDC, ba, sT, xA);
        adj_gemm<<<ag, 128, 0, stream>>>(adjB, sT, ba, xA);
        feat_gemm<<<fg, 128, 0, stream>>>(xA, HIDC, HIDC / 32, wb, HIDC, bb, sT, xB);
        adj_gemm<<<ag, 128, 0, stream>>>(adjB, sT, bb, xB);
        res_acc<<<ew, 256, 0, stream>>>(xB, fbuf);
    }

    // gc13: W_mid[11]
    feat_gemm<<<fg, 128, 0, stream>>>(fbuf, HIDC, HIDC / 32,
                                      WmT + (size_t)11 * HIDC * HIDC, HIDC,
                                      bmid + 11 * HIDC, sT, xA);
    adj_gemm<<<ag, 128, 0, stream>>>(adjB, sT, bmid + 11 * HIDC, xA);
    res_acc<<<ew, 256, 0, stream>>>(xA, fbuf);

    // gc14 (no relu): coords -> d_out[0:24576], feat -> d_out[24576:]
    out_gemm<<<(NND * 3 + 255) / 256, 256, 0, stream>>>(fbuf, Wout, sup3);
    out_agg<<<NND, 256, 0, stream>>>(adjB, sup3, bout, outp);
    feat_out<<<ew, 256, 0, stream>>>(fbuf, outp + (size_t)NND * 3);
}
